// pointnet2_backbone_25967372271716
// MI455X (gfx1250) — matrix operand order repaired and checked by reading, not compiled, not run on hardware
//
#include <hip/hip_runtime.h>
#include <cstdint>
#include <cstddef>

// ---------------- types / helpers ----------------
typedef unsigned short bfbits;
typedef __attribute__((ext_vector_type(16))) __bf16 v16bf;
typedef __attribute__((ext_vector_type(8)))  float  v8f;
typedef __attribute__((ext_vector_type(8)))  unsigned short us8;

__device__ __forceinline__ bfbits f2bf(float f) {
  unsigned u = __float_as_uint(f);
  u += 0x7FFFu + ((u >> 16) & 1u);   // round-to-nearest-even
  return (bfbits)(u >> 16);
}
__device__ __forceinline__ float bf2f(bfbits h) {
  return __uint_as_float(((unsigned)h) << 16);
}
static inline int pad32(int x) { return (x + 31) & ~31; }

// ---------------- [B,3,N] -> [B,N,3] ----------------
__global__ void k_transpose(const float* __restrict__ in, float* __restrict__ out, int B, int N) {
  int i = blockIdx.x * blockDim.x + threadIdx.x;
  if (i >= B * N) return;
  int b = i / N, n = i % N;
  const float* src = in + (size_t)b * 3 * N + n;
  float* dst = out + (size_t)i * 3;
  dst[0] = src[0];
  dst[1] = src[(size_t)N];
  dst[2] = src[(size_t)2 * N];
}

// ---------------- W[Cin,Cout] f32 -> Wt[Np,Kp] bf16 (transposed, zero padded) ----------------
__global__ void k_prep_w(const float* __restrict__ W, bfbits* __restrict__ Wt,
                         int Cin, int Cout, int Kp, int Np) {
  int i = blockIdx.x * blockDim.x + threadIdx.x;
  if (i >= Np * Kp) return;
  int n = i / Kp, k = i % Kp;
  float v = (n < Cout && k < Cin) ? W[(size_t)k * Cout + n] : 0.0f;
  Wt[i] = f2bf(v);
}

// ---------------- farthest point sampling (one block per batch, dist in LDS) ----------------
__global__ void k_fps(const float* __restrict__ xyz, int* __restrict__ idx, int N, int npoint) {
  __shared__ float s_dist[8192];
  __shared__ float s_rv[512];
  __shared__ int   s_ri[512];
  int b = blockIdx.x;
  const float* px = xyz + (size_t)b * N * 3;
  for (int i = threadIdx.x; i < N; i += blockDim.x) s_dist[i] = 1e10f;
  __syncthreads();
  int cur = 0;
  for (int it = 0; it < npoint; ++it) {
    if (threadIdx.x == 0) idx[b * npoint + it] = cur;
    float cx = px[cur * 3 + 0], cy = px[cur * 3 + 1], cz = px[cur * 3 + 2];
    float best = -1.0f; int besti = 0;
    for (int i = threadIdx.x; i < N; i += blockDim.x) {
      float dx = px[i * 3 + 0] - cx, dy = px[i * 3 + 1] - cy, dz = px[i * 3 + 2] - cz;
      float d = dx * dx + dy * dy + dz * dz;
      float dm = fminf(s_dist[i], d);
      s_dist[i] = dm;
      if (dm > best || (dm == best && i < besti)) { best = dm; besti = i; }
    }
    s_rv[threadIdx.x] = best; s_ri[threadIdx.x] = besti;
    __syncthreads();
    for (int s = blockDim.x >> 1; s > 0; s >>= 1) {
      if ((int)threadIdx.x < s) {
        float ov = s_rv[threadIdx.x + s]; int oi = s_ri[threadIdx.x + s];
        float mv = s_rv[threadIdx.x];     int mi = s_ri[threadIdx.x];
        if (ov > mv || (ov == mv && oi < mi)) { s_rv[threadIdx.x] = ov; s_ri[threadIdx.x] = oi; }
      }
      __syncthreads();
    }
    cur = s_ri[0];
    __syncthreads();
  }
}

// ---------------- gather sampled centers ----------------
__global__ void k_gather(const float* __restrict__ xyz, const int* __restrict__ idx,
                         float* __restrict__ out, int N, int total /*B*S*/, int S) {
  int i = blockIdx.x * blockDim.x + threadIdx.x;
  if (i >= total) return;
  int b = i / S;
  int g = idx[i];
  if (g < 0) g = 0; if (g >= N) g = N - 1;
  const float* p = xyz + ((size_t)b * N + g) * 3;
  out[(size_t)i * 3 + 0] = p[0];
  out[(size_t)i * 3 + 1] = p[1];
  out[(size_t)i * 3 + 2] = p[2];
}

// ---------------- ball query + grouped feature build (one wave per center) ----------------
// feat row layout: [pts(Cp) | gxyz(3) | zero pad] -> bf16 X[(b*S+s)*K + j, Kp]
__global__ void k_qb(const float* __restrict__ xyz, const float* __restrict__ nxyz,
                     const float* __restrict__ pts, bfbits* __restrict__ X,
                     int B, int N, int S, int K, int Cp, int Kp, float r2) {
  int wave = (int)((blockIdx.x * blockDim.x + threadIdx.x) >> 5);
  int lane = threadIdx.x & 31;
  __shared__ int selS[4 * 32];
  if (wave >= B * S) return;                 // wave-uniform
  int* sel = selS + ((threadIdx.x >> 5) << 5);
  int b = wave / S;
  float cx = nxyz[(size_t)wave * 3 + 0];
  float cy = nxyz[(size_t)wave * 3 + 1];
  float cz = nxyz[(size_t)wave * 3 + 2];
  int cnt = 0;
  for (int base = 0; base < N && cnt < K; base += 32) {
    int j = base + lane;
    bool in = false;
    if (j < N) {
      const float* p = xyz + ((size_t)b * N + j) * 3;
      float dx = p[0] - cx, dy = p[1] - cy, dz = p[2] - cz;
      in = (dx * dx + dy * dy + dz * dz) <= r2;
    }
    unsigned m = __builtin_amdgcn_ballot_w32(in);
    int rank = cnt + __popc(m & ((1u << lane) - 1u));
    if (in && rank < K) sel[rank] = j;
    cnt += __popc(m);
  }
  if (cnt > K) cnt = K;
  if (cnt == 0) { if (lane == 0) sel[0] = 0; cnt = 1; }
  int first = sel[0];
  for (int j = 0; j < K; ++j) {
    int gi = (j < cnt) ? sel[j] : first;
    size_t row = ((size_t)wave * K + j) * (size_t)Kp;
    for (int c = lane; c < Kp; c += 32) {
      float v;
      if (c < Cp) {
        v = pts[((size_t)b * N + gi) * Cp + c];
      } else if (c < Cp + 3) {
        int a = c - Cp;
        float ctr = (a == 0) ? cx : ((a == 1) ? cy : cz);
        v = xyz[((size_t)b * N + gi) * 3 + a] - ctr;
      } else v = 0.0f;
      X[row + c] = f2bf(v);
    }
  }
}

// ---------------- GEMM + bias/affine/ReLU epilogue via v_wmma_f32_16x16x32_bf16 ----------------
// X: [M,Kp] bf16 row-major. Wt: [Np,Kp] bf16 (column n of W stored contiguously).
// Yb: [M,Np] bf16 (pad cols zero). Optional Yf f32 (stride Np).
__global__ void __launch_bounds__(128) k_gemm(
    const bfbits* __restrict__ X, const bfbits* __restrict__ Wt,
    const float* __restrict__ bias, const float* __restrict__ gain, const float* __restrict__ beta,
    bfbits* __restrict__ Yb, float* __restrict__ Yf,
    int M, int Kp, int N, int Np)
{
  int lane = threadIdx.x & 31;
  int wid = (int)((blockIdx.x * blockDim.x + threadIdx.x) >> 5);
  int tM = M >> 4, tN = Np >> 4;
  if (wid >= tM * tN) return;                 // wave-uniform
  int tm = wid / tN, tn = wid % tN;

  // A (16x32 bf16): lane<16 -> row M=lane, K chunks [0..7] & [16..23]; lane>=16 -> K chunks [8..15] & [24..31]
  int rowA = (tm << 4) + (lane & 15);
  int cA   = (lane >> 4) << 3;                // 0 or 8
  // B (32x16 bf16): column n=lane&15; lane<16 -> K chunks [0..7] & [16..23]; lane>=16 -> K chunks [8..15] & [24..31]
  int colB = (tn << 4) + (lane & 15);
  int kB   = (lane >> 4) << 3;                // 0 or 8
  const bfbits* pA = X  + (size_t)rowA * Kp + cA;
  const bfbits* pB = Wt + (size_t)colB * Kp + kB;

  v8f acc = {0.f, 0.f, 0.f, 0.f, 0.f, 0.f, 0.f, 0.f};
  for (int k0 = 0; k0 < Kp; k0 += 32) {
    union { us8 h[2]; v16bf v; } a, bm;
    a.h[0]  = *(const us8*)(pA + k0);
    a.h[1]  = *(const us8*)(pA + k0 + 16);
    bm.h[0] = *(const us8*)(pB + k0);
    bm.h[1] = *(const us8*)(pB + k0 + 16);
    acc = __builtin_amdgcn_wmma_f32_16x16x32_bf16(false, a.v, false, bm.v,
                                                  (short)0, acc, false, false);
  }

  int col = colB;
  float bb = 0.f, gg = 0.f, be = 0.f;
  if (col < N) { bb = bias[col]; gg = gain[col]; be = beta[col]; }
#pragma unroll
  for (int i = 0; i < 8; ++i) {
    int row = (tm << 4) + i + ((lane >> 4) << 3);   // C/D layout: vgpr i -> row i (+8 for hi lanes)
    float v = fmaxf((acc[i] + bb) * gg + be, 0.0f);
    if (col >= N) v = 0.0f;
    Yb[(size_t)row * Np + col] = f2bf(v);
    if (Yf) Yf[(size_t)row * Np + col] = v;
  }
}

// ---------------- max over K grouped rows ----------------
__global__ void k_maxk(const bfbits* __restrict__ Y, float* __restrict__ out,
                       int G, int K, int N, int Np, int Cstride, int coff) {
  int i = blockIdx.x * blockDim.x + threadIdx.x;
  if (i >= G * N) return;
  int g = i / N, n = i % N;
  const bfbits* p = Y + (size_t)g * K * Np + n;
  float m = bf2f(p[0]);
  for (int j = 1; j < K; ++j) m = fmaxf(m, bf2f(p[(size_t)j * Np]));
  out[(size_t)g * Cstride + coff + n] = m;
}

// ---------------- sa4 feature build: [xyz(3) | l3(C) | pad] ----------------
__global__ void k_sa4(const float* __restrict__ xyz3, const float* __restrict__ l3,
                      bfbits* __restrict__ X, int rows, int C, int Kp) {
  int i = blockIdx.x * blockDim.x + threadIdx.x;
  if (i >= rows * Kp) return;
  int row = i / Kp, c = i % Kp;
  float v = 0.0f;
  if (c < 3) v = xyz3[(size_t)row * 3 + c];
  else if (c < 3 + C) v = l3[(size_t)row * C + (c - 3)];
  X[i] = f2bf(v);
}

// ---------------- fp4 build: [l3(C1) | broadcast l4(C2)] ----------------
__global__ void k_fp4(const float* __restrict__ l3, const float* __restrict__ l4,
                      bfbits* __restrict__ X, int S, int rows, int C1, int C2, int Kp) {
  int i = blockIdx.x * blockDim.x + threadIdx.x;
  if (i >= rows * Kp) return;
  int row = i / Kp, c = i % Kp;
  int b = row / S;
  float v = 0.0f;
  if (c < C1) v = l3[(size_t)row * C1 + c];
  else if (c < C1 + C2) v = l4[(size_t)b * C2 + (c - C1)];
  X[i] = f2bf(v);
}

// ---------------- 3-NN interpolation + concat build (one wave per target point) ----------------
__global__ void k_knn3(const float* __restrict__ xyz1, const float* __restrict__ xyz2,
                       const float* __restrict__ p1, const float* __restrict__ p2,
                       bfbits* __restrict__ X,
                       int B, int N1, int S, int C1, int C2, int Kp) {
  int wave = (int)((blockIdx.x * blockDim.x + threadIdx.x) >> 5);
  int lane = threadIdx.x & 31;
  __shared__ float dsh[4][96];
  __shared__ int   ish[4][96];
  __shared__ int   oi[4][3];
  __shared__ float ow[4][3];
  if (wave >= B * N1) return;                 // wave-uniform
  int w = threadIdx.x >> 5;
  int b = wave / N1, n = wave % N1;
  float px = xyz1[(size_t)wave * 3 + 0];
  float py = xyz1[(size_t)wave * 3 + 1];
  float pz = xyz1[(size_t)wave * 3 + 2];
  float b0 = 1e30f, b1 = 1e30f, b2 = 1e30f; int i0 = 0, i1 = 0, i2 = 0;
  for (int j = lane; j < S; j += 32) {
    const float* q = xyz2 + ((size_t)b * S + j) * 3;
    float dx = q[0] - px, dy = q[1] - py, dz = q[2] - pz;
    float d = dx * dx + dy * dy + dz * dz;
    if (d < b0)      { b2 = b1; i2 = i1; b1 = b0; i1 = i0; b0 = d; i0 = j; }
    else if (d < b1) { b2 = b1; i2 = i1; b1 = d; i1 = j; }
    else if (d < b2) { b2 = d; i2 = j; }
  }
  dsh[w][lane * 3 + 0] = b0; ish[w][lane * 3 + 0] = i0;
  dsh[w][lane * 3 + 1] = b1; ish[w][lane * 3 + 1] = i1;
  dsh[w][lane * 3 + 2] = b2; ish[w][lane * 3 + 2] = i2;
  if (lane == 0) {
    float c0 = 1e30f, c1 = 1e30f, c2 = 1e30f; int j0 = 0, j1 = 0, j2 = 0;
    for (int t = 0; t < 96; ++t) {
      float d = dsh[w][t]; int ii = ish[w][t];
      if (d < c0)      { c2 = c1; j2 = j1; c1 = c0; j1 = j0; c0 = d; j0 = ii; }
      else if (d < c1) { c2 = c1; j2 = j1; c1 = d; j1 = ii; }
      else if (d < c2) { c2 = d; j2 = ii; }
    }
    float w0 = 1.0f / (c0 + 1e-8f), w1 = 1.0f / (c1 + 1e-8f), w2 = 1.0f / (c2 + 1e-8f);
    float ws = w0 + w1 + w2;
    oi[w][0] = j0; oi[w][1] = j1; oi[w][2] = j2;
    ow[w][0] = w0 / ws; ow[w][1] = w1 / ws; ow[w][2] = w2 / ws;
  }
  int j0 = oi[w][0], j1 = oi[w][1], j2 = oi[w][2];
  float w0 = ow[w][0], w1 = ow[w][1], w2 = ow[w][2];
  for (int c = lane; c < Kp; c += 32) {
    float v;
    if (c < C1) {
      v = p1[((size_t)b * N1 + n) * C1 + c];
    } else if (c < C1 + C2) {
      int cc = c - C1;
      v = w0 * p2[((size_t)b * S + j0) * C2 + cc]
        + w1 * p2[((size_t)b * S + j1) * C2 + cc]
        + w2 * p2[((size_t)b * S + j2) * C2 + cc];
    } else v = 0.0f;
    X[(size_t)wave * Kp + c] = f2bf(v);
  }
}

__global__ void k_copy(const float* __restrict__ src, float* __restrict__ dst, int n) {
  int i = blockIdx.x * blockDim.x + threadIdx.x;
  if (i < n) dst[i] = src[i];
}

// ---------------- layer tables ----------------
struct LD { int cin, cout; };
static const LD LAYERS[30] = {
  {3,16},{16,16},{16,32},        // 0-2  sa1 b0
  {3,32},{32,32},{32,64},        // 3-5  sa1 b1
  {99,64},{64,64},{64,128},      // 6-8  sa2 b0
  {99,64},{64,96},{96,128},      // 9-11 sa2 b1
  {259,128},{128,196},{196,256}, // 12-14 sa3 b0
  {259,128},{128,196},{196,256}, // 15-17 sa3 b1
  {515,256},{256,512},{512,512}, // 18-20 sa4
  {1024,256},{256,256},          // 21-22 fp4
  {512,256},{256,256},           // 23-24 fp3
  {352,256},{256,128},           // 25-26 fp2
  {128,128},{128,128},{128,128}  // 27-29 fp1
};
// input order -> struct layer index
static const int ORDER_SORTED[30] = {27,28,29, 25,26, 23,24, 21,22,
                                     0,1,2,3,4,5, 6,7,8,9,10,11,
                                     12,13,14,15,16,17, 18,19,20};
static const int ORDER_INSERT[30] = {0,1,2,3,4,5, 6,7,8,9,10,11,
                                     12,13,14,15,16,17, 18,19,20,
                                     21,22, 23,24, 25,26, 27,28,29};

static inline unsigned nblk(long long total, int bs) {
  return (unsigned)((total + bs - 1) / bs);
}

extern "C" void kernel_launch(void* const* d_in, const int* in_sizes, int n_in,
                              void* d_out, int out_size, void* d_ws, size_t ws_size,
                              hipStream_t stream) {
  const int B = 8, N0 = 8192;
  // locate xyz by unique size
  int xyzIdx = -1;
  for (int i = 0; i < n_in; ++i) if (in_sizes[i] == B * 3 * N0) { xyzIdx = i; break; }
  if (xyzIdx < 0 || n_in < 121) return;
  int pidx[120]; int npar = 0;
  for (int i = 0; i < n_in && npar < 120; ++i) if (i != xyzIdx) pidx[npar++] = i;
  if (npar < 120) return;

  auto matches = [&](const int* ord) -> bool {
    for (int j = 0; j < 30; ++j) {
      const LD& L = LAYERS[ord[j]];
      if (in_sizes[pidx[4*j+0]] != L.cin * L.cout) return false;
      if (in_sizes[pidx[4*j+1]] != L.cout) return false;
      if (in_sizes[pidx[4*j+2]] != L.cout) return false;
      if (in_sizes[pidx[4*j+3]] != L.cout) return false;
    }
    return true;
  };
  const int* ord = ORDER_SORTED;
  if (!matches(ord)) { if (matches(ORDER_INSERT)) ord = ORDER_INSERT; }

  const float *Wp[30], *bp[30], *gp[30], *ep[30];
  for (int j = 0; j < 30; ++j) {
    int s = ord[j];
    Wp[s] = (const float*)d_in[pidx[4*j+0]];
    bp[s] = (const float*)d_in[pidx[4*j+1]];
    gp[s] = (const float*)d_in[pidx[4*j+2]];
    ep[s] = (const float*)d_in[pidx[4*j+3]];
  }

  // ---- workspace bump allocator ----
  char* base = (char*)d_ws;
  size_t off = 0;
  auto alloc = [&](size_t bytes) -> void* {
    off = (off + 255) & ~(size_t)255;
    void* p = base + off;
    off += bytes;
    return p;
  };
  float* l0x = (float*)alloc((size_t)B * N0 * 3 * 4);
  float* l1x = (float*)alloc((size_t)B * 1024 * 3 * 4);
  float* l2x = (float*)alloc((size_t)B * 256 * 3 * 4);
  float* l3x = (float*)alloc((size_t)B * 64 * 3 * 4);
  int*   idxb = (int*)alloc((size_t)B * 1024 * 4);
  float* l1  = (float*)alloc((size_t)B * 1024 * 96 * 4);
  float* l2  = (float*)alloc((size_t)B * 256 * 256 * 4);
  float* l3  = (float*)alloc((size_t)B * 64 * 512 * 4);
  float* l4  = (float*)alloc((size_t)B * 512 * 4);
  float* l3p = (float*)alloc((size_t)B * 64 * 256 * 4);
  float* l2p = (float*)alloc((size_t)B * 256 * 256 * 4);
  float* l1p = (float*)alloc((size_t)B * 1024 * 128 * 4);
  bfbits* Wt[30];
  for (int s = 0; s < 30; ++s) {
    int Kp = pad32(LAYERS[s].cin), Np = pad32(LAYERS[s].cout);
    Wt[s] = (bfbits*)alloc((size_t)Np * Kp * 2);
  }
  bfbits* xbuf = (bfbits*)alloc((size_t)8388608 * 2);   // max M*Kp
  bfbits* actA = (bfbits*)alloc((size_t)16777216 * 2);  // max M*Np (sa1 b1 L2)
  bfbits* actB = (bfbits*)alloc((size_t)8388608 * 2);   // max odd-layer M*Np
  if (off > ws_size) return;

  // ---- weight prep + input transpose ----
  k_transpose<<<nblk((long long)B * N0, 256), 256, 0, stream>>>((const float*)d_in[xyzIdx], l0x, B, N0);
  for (int s = 0; s < 30; ++s) {
    int Kp = pad32(LAYERS[s].cin), Np = pad32(LAYERS[s].cout);
    k_prep_w<<<nblk((long long)Np * Kp, 256), 256, 0, stream>>>(Wp[s], Wt[s], LAYERS[s].cin, LAYERS[s].cout, Kp, Np);
  }

  // ---- MLP chain runner ----
  auto run_chain = [&](int first, int nl, bfbits* X, int M, float* finalYf) -> bfbits* {
    bfbits* in = X;
    for (int l = 0; l < nl; ++l) {
      int s = first + l;
      int Kp = pad32(LAYERS[s].cin);
      int N = LAYERS[s].cout, Np = pad32(N);
      bfbits* out = (l & 1) ? actB : actA;
      float* yf = (l == nl - 1) ? finalYf : nullptr;
      long long waves = (long long)(M / 16) * (Np / 16);
      k_gemm<<<nblk(waves, 4), 128, 0, stream>>>(in, Wt[s], bp[s], gp[s], ep[s], out, yf, M, Kp, N, Np);
      in = out;
    }
    return in;
  };

  // ================= SA1: N=8192 -> S=1024 =================
  k_fps<<<B, 512, 0, stream>>>(l0x, idxb, N0, 1024);
  k_gather<<<nblk((long long)B * 1024, 256), 256, 0, stream>>>(l0x, idxb, l1x, N0, B * 1024, 1024);
  k_qb<<<nblk((long long)B * 1024, 4), 128, 0, stream>>>(l0x, l1x, (const float*)nullptr, xbuf,
                                                         B, N0, 1024, 16, 0, 32, 0.05f * 0.05f);
  { bfbits* f = run_chain(0, 3, xbuf, 131072, nullptr);
    k_maxk<<<nblk((long long)B * 1024 * 32, 256), 256, 0, stream>>>(f, l1, B * 1024, 16, 32, 32, 96, 0); }
  k_qb<<<nblk((long long)B * 1024, 4), 128, 0, stream>>>(l0x, l1x, (const float*)nullptr, xbuf,
                                                         B, N0, 1024, 32, 0, 32, 0.1f * 0.1f);
  { bfbits* f = run_chain(3, 3, xbuf, 262144, nullptr);
    k_maxk<<<nblk((long long)B * 1024 * 64, 256), 256, 0, stream>>>(f, l1, B * 1024, 32, 64, 64, 96, 32); }

  // ================= SA2: 1024 -> 256 =================
  k_fps<<<B, 512, 0, stream>>>(l1x, idxb, 1024, 256);
  k_gather<<<nblk((long long)B * 256, 256), 256, 0, stream>>>(l1x, idxb, l2x, 1024, B * 256, 256);
  k_qb<<<nblk((long long)B * 256, 4), 128, 0, stream>>>(l1x, l2x, l1, xbuf,
                                                        B, 1024, 256, 16, 96, 128, 0.1f * 0.1f);
  { bfbits* f = run_chain(6, 3, xbuf, 32768, nullptr);
    k_maxk<<<nblk((long long)B * 256 * 128, 256), 256, 0, stream>>>(f, l2, B * 256, 16, 128, 128, 256, 0); }
  k_qb<<<nblk((long long)B * 256, 4), 128, 0, stream>>>(l1x, l2x, l1, xbuf,
                                                        B, 1024, 256, 32, 96, 128, 0.2f * 0.2f);
  { bfbits* f = run_chain(9, 3, xbuf, 65536, nullptr);
    k_maxk<<<nblk((long long)B * 256 * 128, 256), 256, 0, stream>>>(f, l2, B * 256, 32, 128, 128, 256, 128); }

  // ================= SA3: 256 -> 64 =================
  k_fps<<<B, 512, 0, stream>>>(l2x, idxb, 256, 64);
  k_gather<<<nblk((long long)B * 64, 256), 256, 0, stream>>>(l2x, idxb, l3x, 256, B * 64, 64);
  k_qb<<<nblk((long long)B * 64, 4), 128, 0, stream>>>(l2x, l3x, l2, xbuf,
                                                       B, 256, 64, 16, 256, 288, 0.2f * 0.2f);
  { bfbits* f = run_chain(12, 3, xbuf, 8192, nullptr);
    k_maxk<<<nblk((long long)B * 64 * 256, 256), 256, 0, stream>>>(f, l3, B * 64, 16, 256, 256, 512, 0); }
  k_qb<<<nblk((long long)B * 64, 4), 128, 0, stream>>>(l2x, l3x, l2, xbuf,
                                                       B, 256, 64, 32, 256, 288, 0.4f * 0.4f);
  { bfbits* f = run_chain(15, 3, xbuf, 16384, nullptr);
    k_maxk<<<nblk((long long)B * 64 * 256, 256), 256, 0, stream>>>(f, l3, B * 64, 32, 256, 256, 512, 256); }

  // ================= SA4 (global): feat=[xyz|l3], max over 64 =================
  k_sa4<<<nblk((long long)512 * 544, 256), 256, 0, stream>>>(l3x, l3, xbuf, 512, 512, 544);
  { bfbits* f = run_chain(18, 3, xbuf, 512, nullptr);
    k_maxk<<<nblk((long long)B * 512, 256), 256, 0, stream>>>(f, l4, B, 64, 512, 512, 512, 0); }
  k_copy<<<nblk(4096, 256), 256, 0, stream>>>(l4, (float*)d_out, 4096);

  // ================= FP4: broadcast l4 to 64 pts, concat l3 =================
  k_fp4<<<nblk((long long)512 * 1024, 256), 256, 0, stream>>>(l3, l4, xbuf, 64, 512, 512, 512, 1024);
  run_chain(21, 2, xbuf, 512, l3p);

  // ================= FP3: interp (64->256), concat l2 =================
  k_knn3<<<nblk((long long)B * 256, 4), 128, 0, stream>>>(l2x, l3x, l2, l3p, xbuf,
                                                          B, 256, 64, 256, 256, 512);
  run_chain(23, 2, xbuf, 2048, l2p);

  // ================= FP2: interp (256->1024), concat l1 =================
  k_knn3<<<nblk((long long)B * 1024, 4), 128, 0, stream>>>(l1x, l2x, l1, l2p, xbuf,
                                                           B, 1024, 256, 96, 256, 352);
  run_chain(25, 2, xbuf, 8192, l1p);

  // ================= FP1: interp (1024->8192), final point features =================
  k_knn3<<<nblk((long long)B * N0, 4), 128, 0, stream>>>(l0x, l1x, (const float*)nullptr, l1p, xbuf,
                                                         B, N0, 1024, 0, 128, 128);
  run_chain(27, 3, xbuf, 65536, (float*)d_out + 4096);
}
